// BoundaryNet_20890720928298
// MI455X (gfx1250) — matrix operand order repaired and checked by reading, not compiled, not run on hardware
//
#include <hip/hip_runtime.h>
#include <hip/hip_bf16.h>

typedef __attribute__((ext_vector_type(16))) _Float16 v16h;
typedef __attribute__((ext_vector_type(8)))  float    v8f;

#define TPB 256
static inline dim3 GRD(long n, int b) { return dim3((unsigned)((n + b - 1) / b)); }

// ---------------------------------------------------------------------------
// Generic GEMM: C(MxN) = A(MxK) @ B(KxN) [+ bias per column]
// f32 in/out, f16 WMMA compute with f32 accumulate.
// One wave computes a 64x16 strip of C: 4 M-tiles sharing one B fragment per
// K-step. Path selection (interior vs clamped-tail) is hoisted OUT of the
// K-loop so accumulators stay pinned (no phi copies). Tail-step zero-fill is
// done with a mask-multiply on unconditionally loaded (clamped, in-bounds)
// values so the compiler cannot sink loads under exec-mask branches.
// Fragments are built directly from global memory in the CDNA5 wave32 layout:
//   A frag: a[h] = A[row][Kmap],  Kmap = kt + h + half*8        (h < 8)
//                                 kt + 16 + (h-8) + half*8      (h >= 8)
//   B frag: b[h] = B[Kmap][colBase + lane&15]   (same Kmap as A)
// ---------------------------------------------------------------------------
__device__ __forceinline__
void gemm_step_fast(const float* __restrict__ A, const float* __restrict__ B,
                    int N, int K, int rowBase, int colBase, int kt,
                    int half, int l15, v8f acc[4])
{
    const float* __restrict__ Bcol = B + (size_t)kt * N + colBase + l15;
    v16h b;
#pragma unroll
    for (int h = 0; h < 16; ++h)
        b[h] = (_Float16)Bcol[(size_t)((h < 8 ? h : h + 8) + half * 8) * N];

#pragma unroll
    for (int t = 0; t < 4; ++t) {
        const float* __restrict__ Arow =
            A + (size_t)(rowBase + t * 16 + l15) * K + kt + half * 8;
        if (t == 0 && kt + 32 < K)   // speculative prefetch of next A K-slab
            __builtin_prefetch(Arow + 32, 0, 0);
        v16h a;
#pragma unroll
        for (int h = 0; h < 8; ++h) {
            a[h]     = (_Float16)Arow[h];
            a[h + 8] = (_Float16)Arow[h + 16];
        }
        acc[t] = __builtin_amdgcn_wmma_f32_16x16x32_f16(false, a, false, b,
                                                        (short)0, acc[t], false, false);
    }
}

__device__ __forceinline__
void gemm_step_tail(const float* __restrict__ A, const float* __restrict__ B,
                    int M, int N, int K, int rowBase, int colBase, int kt,
                    int half, int l15, v8f acc[4])
{
    int gc = colBase + l15;
    int ac = gc < N ? gc : N - 1;
    float cm = (gc < N) ? 1.0f : 0.0f;

    float tb[16];
#pragma unroll
    for (int h = 0; h < 16; ++h) {
        int gkb = kt + (h < 8 ? h : h + 8) + half * 8;
        int akb = gkb < K ? gkb : K - 1;
        tb[h] = B[(size_t)akb * N + ac];          // clamped: always in-bounds
    }
    v16h b;
#pragma unroll
    for (int h = 0; h < 16; ++h) {
        int gkb = kt + (h < 8 ? h : h + 8) + half * 8;
        b[h] = (_Float16)(tb[h] * ((gkb < K) ? cm : 0.0f));
    }

#pragma unroll
    for (int t = 0; t < 4; ++t) {
        int gr = rowBase + t * 16 + l15;
        int ar = gr < M ? gr : M - 1;
        float rm = (gr < M) ? 1.0f : 0.0f;
        float ta[16];
#pragma unroll
        for (int h = 0; h < 16; ++h) {
            int gk = kt + (h < 8 ? h : h + 8) + half * 8;
            int ak = gk < K ? gk : K - 1;
            ta[h] = A[(size_t)ar * K + ak];       // clamped: always in-bounds
        }
        v16h a;
#pragma unroll
        for (int h = 0; h < 16; ++h) {
            int gk = kt + (h < 8 ? h : h + 8) + half * 8;
            a[h] = (_Float16)(ta[h] * ((gk < K) ? rm : 0.0f));
        }
        acc[t] = __builtin_amdgcn_wmma_f32_16x16x32_f16(false, a, false, b,
                                                        (short)0, acc[t], false, false);
    }
}

__global__ __launch_bounds__(32)
void gemm_wmma_kernel(const float* __restrict__ A,
                      const float* __restrict__ B,
                      const float* __restrict__ bias,
                      float* __restrict__ C,
                      int M, int N, int K)
{
    const int lane = threadIdx.x;
    const int half = lane >> 4;       // 0: lanes 0-15, 1: lanes 16-31
    const int l15  = lane & 15;
    const int rowBase = blockIdx.x * 64;
    const int colBase = blockIdx.y * 16;

    const bool interior = (rowBase + 64 <= M) && (colBase + 16 <= N); // wave-uniform

    v8f acc[4] = {};

    if (interior) {
        const int kFull = K & ~31;
        for (int kt = 0; kt < kFull; kt += 32)
            gemm_step_fast(A, B, N, K, rowBase, colBase, kt, half, l15, acc);
        if (kFull < K)
            gemm_step_tail(A, B, M, N, K, rowBase, colBase, kFull, half, l15, acc);
    } else {
        for (int kt = 0; kt < K; kt += 32)
            gemm_step_tail(A, B, M, N, K, rowBase, colBase, kt, half, l15, acc);
    }

#pragma unroll
    for (int t = 0; t < 4; ++t) {
#pragma unroll
        for (int r = 0; r < 8; ++r) {
            int gr = rowBase + t * 16 + r + half * 8;
            int gc = colBase + l15;
            if (gr < M && gc < N) {
                float v = acc[t][r];
                if (bias) v += bias[gc];
                C[(size_t)gr * N + gc] = v;
            }
        }
    }
}

static void gemm(const float* A, const float* B, const float* bias, float* C,
                 int M, int N, int K, hipStream_t s)
{
    dim3 grid((M + 63) / 64, (N + 15) / 16);
    gemm_wmma_kernel<<<grid, 32, 0, s>>>(A, B, bias, C, M, N, K);
}

// ---------------------------------------------------------------------------
// BatchNorm (training-mode, biased variance) helpers
// ---------------------------------------------------------------------------
__global__ void bn_stats_kernel(const float* __restrict__ X, int M, int N,
                                float* __restrict__ mean, float* __restrict__ rstd)
{
    int col = blockIdx.x, tid = threadIdx.x;
    float s = 0.0f, s2 = 0.0f;
    for (int r = tid; r < M; r += 256) {
        float v = X[(size_t)r * N + col];
        s += v; s2 += v * v;
    }
    __shared__ float sh[256], sh2[256];
    sh[tid] = s; sh2[tid] = s2;
    __syncthreads();
    for (int st = 128; st > 0; st >>= 1) {
        if (tid < st) { sh[tid] += sh[tid + st]; sh2[tid] += sh2[tid + st]; }
        __syncthreads();
    }
    if (tid == 0) {
        float m = sh[0] / (float)M;
        float var = sh2[0] / (float)M - m * m;
        mean[col] = m;
        rstd[col] = rsqrtf(var + 1e-5f);
    }
}

__global__ void bn_relu_kernel(const float* __restrict__ X, float* __restrict__ Y,
                               int total, int N,
                               const float* __restrict__ mean, const float* __restrict__ rstd,
                               const float* __restrict__ g, const float* __restrict__ b)
{
    int e = blockIdx.x * blockDim.x + threadIdx.x;
    if (e >= total) return;
    int c = e % N;
    float v = (X[e] - mean[c]) * rstd[c] * g[c] + b[c];
    Y[e] = v > 0.0f ? v : 0.0f;
}

// relu(bn(.)) followed by max over the 16-sample neighborhood (fused)
__global__ void bn_relu_max_kernel(const float* __restrict__ X, float* __restrict__ Y,
                                   int m, int N,
                                   const float* __restrict__ mean, const float* __restrict__ rstd,
                                   const float* __restrict__ g, const float* __restrict__ b)
{
    int e = blockIdx.x * blockDim.x + threadIdx.x;
    if (e >= m * N) return;
    int c = e % N, mi = e / N;
    float gg = g[c], bb = b[c], mm = mean[c], rs = rstd[c];
    float mx = 0.0f;   // relu output is >= 0, so max(relu(v)) == max(0, max(v))
#pragma unroll 4
    for (int j = 0; j < 16; ++j) {
        float v = (X[(size_t)(mi * 16 + j) * N + c] - mm) * rs * gg + bb;
        if (v > mx) mx = v;
    }
    Y[e] = mx;
}

// ---------------------------------------------------------------------------
// Furthest point sampling: single workgroup, sequential argmax loop
// ---------------------------------------------------------------------------
__global__ void fps_kernel(const float* __restrict__ p, int n, int m,
                           int* __restrict__ idxs, float* __restrict__ dists)
{
    const int tid = threadIdx.x;
    __shared__ float red_v[1024];
    __shared__ int   red_i[1024];
    __shared__ float lastp[3];

    for (int j = tid; j < n; j += 1024) dists[j] = 1e10f;
    if (tid == 0) idxs[0] = 0;
    __syncthreads();

    for (int i = 0; i < m - 1; ++i) {
        if (tid == 0) {
            int li = idxs[i];
            lastp[0] = p[li * 3 + 0];
            lastp[1] = p[li * 3 + 1];
            lastp[2] = p[li * 3 + 2];
        }
        __syncthreads();
        float lx = lastp[0], ly = lastp[1], lz = lastp[2];
        float bv = -1.0f; int bi = 0;
        for (int j = tid; j < n; j += 1024) {
            float dx = p[j * 3 + 0] - lx;
            float dy = p[j * 3 + 1] - ly;
            float dz = p[j * 3 + 2] - lz;
            float d = dx * dx + dy * dy + dz * dz;
            float nd = fminf(dists[j], d);
            dists[j] = nd;
            if (nd > bv) { bv = nd; bi = j; }
        }
        red_v[tid] = bv; red_i[tid] = bi;
        __syncthreads();
        for (int s = 512; s > 0; s >>= 1) {
            if (tid < s) {
                float ov = red_v[tid + s]; int oi = red_i[tid + s];
                if (ov > red_v[tid] || (ov == red_v[tid] && oi < red_i[tid])) {
                    red_v[tid] = ov; red_i[tid] = oi;
                }
            }
            __syncthreads();
        }
        if (tid == 0) idxs[i + 1] = red_i[0];
        __syncthreads();
    }
}

// ---------------------------------------------------------------------------
// KNN: per-thread insertion sort (k nearest, ascending, first-index ties)
// ---------------------------------------------------------------------------
template <int KK>
__global__ void knn_kernel(const float* __restrict__ q, int nq,
                           const float* __restrict__ s, int ns,
                           int* __restrict__ out)
{
    int i = blockIdx.x * blockDim.x + threadIdx.x;
    if (i >= nq) return;
    float qx = q[i * 3 + 0], qy = q[i * 3 + 1], qz = q[i * 3 + 2];
    float bd[KK]; int bi[KK];
#pragma unroll
    for (int t = 0; t < KK; ++t) { bd[t] = 3.4e38f; bi[t] = 0; }
    for (int j = 0; j < ns; ++j) {
        float dx = qx - s[j * 3 + 0];
        float dy = qy - s[j * 3 + 1];
        float dz = qz - s[j * 3 + 2];
        float d = dx * dx + dy * dy + dz * dz;
        if (d < bd[KK - 1]) {
            int pos = KK - 1;
            while (pos > 0 && bd[pos - 1] > d) {
                bd[pos] = bd[pos - 1]; bi[pos] = bi[pos - 1]; --pos;
            }
            bd[pos] = d; bi[pos] = j;
        }
    }
#pragma unroll
    for (int t = 0; t < KK; ++t) out[i * KK + t] = bi[t];
}

// ---------------------------------------------------------------------------
// Gather / concat / interpolation kernels
// ---------------------------------------------------------------------------
__global__ void concat6_kernel(const float* __restrict__ p, const float* __restrict__ x,
                               float* __restrict__ out, int n)
{
    int e = blockIdx.x * blockDim.x + threadIdx.x;
    if (e >= n * 6) return;
    int i = e / 6, c = e % 6;
    out[e] = (c < 3) ? p[i * 3 + c] : x[i * 3 + c - 3];
}

__global__ void sample_kernel(const float* __restrict__ p, const int* __restrict__ fi,
                              float* __restrict__ out, int m)
{
    int e = blockIdx.x * blockDim.x + threadIdx.x;
    if (e >= m * 3) return;
    int i = e / 3, c = e % 3;
    out[e] = p[fi[i] * 3 + c];
}

// G[(mi*16+j)][c] = c<3 ? p[kn]-np_ : x[kn][c-3]
__global__ void gather_kernel(const float* __restrict__ p, const float* __restrict__ x,
                              int C, const int* __restrict__ kn,
                              const float* __restrict__ np_, float* __restrict__ G, int m)
{
    int F = 3 + C;
    int total = m * 16 * F;
    int e = blockIdx.x * blockDim.x + threadIdx.x;
    if (e >= total) return;
    int c = e % F;
    int rest = e / F;
    int j = rest % 16, mi = rest / 16;
    int src = kn[mi * 16 + j];
    G[e] = (c < 3) ? (p[src * 3 + c] - np_[mi * 3 + c])
                   : x[(size_t)src * C + (c - 3)];
}

// out = y + inverse-squared-distance 3-NN interpolation of z
__global__ void interp_add_kernel(const float* __restrict__ pt, const float* __restrict__ ps,
                                  const int* __restrict__ idx,
                                  const float* __restrict__ z, const float* __restrict__ y,
                                  float* __restrict__ out, int nt, int C)
{
    int e = blockIdx.x * blockDim.x + threadIdx.x;
    if (e >= nt * C) return;
    int i = e / C, c = e % C;
    int i0 = idx[i * 3 + 0], i1 = idx[i * 3 + 1], i2 = idx[i * 3 + 2];
    float px = pt[i * 3 + 0], py = pt[i * 3 + 1], pz = pt[i * 3 + 2];
    float dx, dy, dz;
    dx = px - ps[i0 * 3]; dy = py - ps[i0 * 3 + 1]; dz = pz - ps[i0 * 3 + 2];
    float w0 = 1.0f / (dx * dx + dy * dy + dz * dz + 1e-8f);
    dx = px - ps[i1 * 3]; dy = py - ps[i1 * 3 + 1]; dz = pz - ps[i1 * 3 + 2];
    float w1 = 1.0f / (dx * dx + dy * dy + dz * dz + 1e-8f);
    dx = px - ps[i2 * 3]; dy = py - ps[i2 * 3 + 1]; dz = pz - ps[i2 * 3 + 2];
    float w2 = 1.0f / (dx * dx + dy * dy + dz * dz + 1e-8f);
    float sw = w0 + w1 + w2;
    out[e] = y[e] + (w0 * z[(size_t)i0 * C + c] +
                     w1 * z[(size_t)i1 * C + c] +
                     w2 * z[(size_t)i2 * C + c]) / sw;
}

// ---------------------------------------------------------------------------
// Small vector ops for dec5 global-feature path and the padded boundary head
// ---------------------------------------------------------------------------
__global__ void meanrows_kernel(const float* __restrict__ X, int M, int N,
                                float* __restrict__ out)
{
    int n = blockIdx.x * blockDim.x + threadIdx.x;
    if (n >= N) return;
    float s = 0.0f;
    for (int r = 0; r < M; ++r) s += X[(size_t)r * N + n];
    out[n] = s / (float)M;
}

__global__ void matvec_kernel(const float* __restrict__ vin, const float* __restrict__ W,
                              const float* __restrict__ bias, float* __restrict__ out,
                              int K, int N, int doRelu)
{
    int n = blockIdx.x * blockDim.x + threadIdx.x;
    if (n >= N) return;
    float s = bias ? bias[n] : 0.0f;
    for (int k = 0; k < K; ++k) s += vin[k] * W[(size_t)k * N + n];
    if (doRelu && s < 0.0f) s = 0.0f;
    out[n] = s;
}

// pad bd_W2 (32x2) -> (32x16) and bd_b2 (2) -> (16) with zeros
__global__ void pad_head_kernel(const float* __restrict__ W, const float* __restrict__ bias,
                                float* __restrict__ Wpad, float* __restrict__ bpad)
{
    int e = blockIdx.x * blockDim.x + threadIdx.x;
    if (e < 512) {
        int k = e >> 4, n = e & 15;
        Wpad[e] = (n < 2) ? W[k * 2 + n] : 0.0f;
    } else if (e < 528) {
        int n = e - 512;
        bpad[n] = (n < 2) ? bias[n] : 0.0f;
    }
}

// extract the 2 logit columns from the N=16 padded output
__global__ void extract2_kernel(const float* __restrict__ Xpad, float* __restrict__ out, int n1)
{
    int e = blockIdx.x * blockDim.x + threadIdx.x;
    if (e >= n1 * 2) return;
    int i = e >> 1, c = e & 1;
    out[e] = Xpad[(size_t)i * 16 + c];
}

// ---------------------------------------------------------------------------
// Host orchestration
// ---------------------------------------------------------------------------
extern "C" void kernel_launch(void* const* d_in, const int* in_sizes, int n_in,
                              void* d_out, int out_size, void* d_ws, size_t ws_size,
                              hipStream_t stream)
{
    (void)in_sizes; (void)n_in; (void)out_size; (void)ws_size;

    const float* p0 = (const float*)d_in[0];
    const float* x0 = (const float*)d_in[1];
    // d_in[2] = o0 (single-segment offsets) unused
    auto prm = [&](int i) -> const float* { return (const float*)d_in[3 + i]; };
    // param indices: enc{l}_W=3*(l-1), _g=+1, _b=+2 for l=1..5
    // 15..18 dec5_l1_{W,b,g,bt}; 19,20 dec5_l2_{W,b}
    // decoder lvl (4..1): base = 21+(4-lvl)*8 -> l1_W,l1_b,l1_g,l1_bt,l2_W,l2_b,l2_g,l2_bt
    // 53..58 db_{W1,b1,g,bt,W2,b2}; 59..64 bd_{W1,b1,g,bt,W2,b2}

    const int cnt[5] = {16384, 4096, 1024, 256, 64};
    const int wid[5] = {32, 64, 128, 256, 512};

    // workspace bump allocator
    char* base = (char*)d_ws;
    size_t off = 0;
    auto allocF = [&](size_t n) -> float* {
        float* p = (float*)(base + off);
        off += ((n * sizeof(float) + 255) / 256) * 256;
        return p;
    };
    auto allocI = [&](size_t n) -> int* {
        int* p = (int*)(base + off);
        off += ((n * sizeof(int) + 255) / 256) * 256;
        return p;
    };

    float* xin   = allocF(16384 * 6);
    float* p2    = allocF(4096 * 3);
    float* p3    = allocF(1024 * 3);
    float* p4    = allocF(256 * 3);
    float* p5    = allocF(64 * 3);
    float* dists = allocF(16384);
    int* fiA[4] = {allocI(4096), allocI(1024), allocI(256), allocI(64)};
    int* knA[4] = {allocI(4096 * 16), allocI(1024 * 16), allocI(256 * 16), allocI(64 * 16)};
    int* itA[4] = {allocI(16384 * 3), allocI(4096 * 3), allocI(1024 * 3), allocI(256 * 3)};
    float* x1 = allocF((size_t)16384 * 32);
    float* x2 = allocF((size_t)4096 * 64);
    float* x3 = allocF((size_t)1024 * 128);
    float* x4 = allocF((size_t)256 * 256);
    float* x5 = allocF((size_t)64 * 512);
    float* G      = allocF((size_t)65536 * 35);   // largest grouped matrix (level 2)
    float* bufPre = allocF((size_t)65536 * 64);   // largest GEMM output
    float* bufY   = allocF((size_t)16384 * 32);
    float* bufZ   = allocF((size_t)4096 * 32);
    float* xbB[5]; // xb5..xb1
    xbB[0] = allocF((size_t)64 * 512);
    xbB[1] = allocF((size_t)256 * 256);
    xbB[2] = allocF((size_t)1024 * 128);
    xbB[3] = allocF((size_t)4096 * 64);
    xbB[4] = allocF((size_t)16384 * 32);
    float* h1 = allocF((size_t)16384 * 32);
    float* h2 = allocF((size_t)16384 * 32);
    float* outPad = allocF((size_t)16384 * 16);
    float* Wpad  = allocF(512);
    float* bpad  = allocF(16);
    float* mvec  = allocF(512);
    float* gfeat = allocF(512);
    float* vvec  = allocF(512);
    float* meanB = allocF(512);
    float* rstdB = allocF(512);

    const float* pts[5] = {p0, p2, p3, p4, p5};
    float* ptsm[5] = {nullptr, p2, p3, p4, p5};
    float* feats[5] = {x1, x2, x3, x4, x5};

    // ---- encoder level 1 ----
    concat6_kernel<<<GRD(16384 * 6, TPB), TPB, 0, stream>>>(p0, x0, xin, 16384);
    gemm(xin, prm(0), nullptr, bufPre, 16384, 32, 6, stream);
    bn_stats_kernel<<<32, 256, 0, stream>>>(bufPre, 16384, 32, meanB, rstdB);
    bn_relu_kernel<<<GRD((size_t)16384 * 32, TPB), TPB, 0, stream>>>(
        bufPre, x1, 16384 * 32, 32, meanB, rstdB, prm(1), prm(2));

    // ---- encoder levels 2..5 (transition down) ----
    for (int l = 0; l < 4; ++l) {
        int nprev = cnt[l], m = cnt[l + 1];
        int Cin = wid[l], Cout = wid[l + 1];
        int wi = 3 * (l + 1); // enc{l+2}_W index

        fps_kernel<<<1, 1024, 0, stream>>>(pts[l], nprev, m, fiA[l], dists);
        sample_kernel<<<GRD(m * 3, TPB), TPB, 0, stream>>>(pts[l], fiA[l], ptsm[l + 1], m);
        knn_kernel<16><<<GRD(m, 128), 128, 0, stream>>>(pts[l + 1], m, pts[l], nprev, knA[l]);
        gather_kernel<<<GRD((size_t)m * 16 * (3 + Cin), TPB), TPB, 0, stream>>>(
            pts[l], feats[l], Cin, knA[l], pts[l + 1], G, m);
        gemm(G, prm(wi), nullptr, bufPre, m * 16, Cout, 3 + Cin, stream);
        bn_stats_kernel<<<Cout, 256, 0, stream>>>(bufPre, m * 16, Cout, meanB, rstdB);
        bn_relu_max_kernel<<<GRD((size_t)m * Cout, TPB), TPB, 0, stream>>>(
            bufPre, feats[l + 1], m, Cout, meanB, rstdB, prm(wi + 1), prm(wi + 2));
    }

    // ---- interpolation caches (3-NN of level t-1 onto level t) ----
    for (int t = 1; t <= 4; ++t)
        knn_kernel<3><<<GRD(cnt[t - 1], 128), 128, 0, stream>>>(
            pts[t - 1], cnt[t - 1], pts[t], cnt[t], itA[t - 1]);

    // ---- dec5 head ----
    meanrows_kernel<<<GRD(512, TPB), TPB, 0, stream>>>(x5, 64, 512, mvec);
    matvec_kernel<<<GRD(512, TPB), TPB, 0, stream>>>(mvec, prm(19), prm(20), gfeat, 512, 512, 1);
    matvec_kernel<<<GRD(512, TPB), TPB, 0, stream>>>(gfeat, prm(15) + (size_t)512 * 512,
                                                     prm(16), vvec, 512, 512, 0);
    gemm(x5, prm(15), vvec, bufPre, 64, 512, 512, stream);   // x5 @ W_top + (gfeat @ W_bot + b)
    bn_stats_kernel<<<512, 256, 0, stream>>>(bufPre, 64, 512, meanB, rstdB);
    bn_relu_kernel<<<GRD((size_t)64 * 512, TPB), TPB, 0, stream>>>(
        bufPre, xbB[0], 64 * 512, 512, meanB, rstdB, prm(17), prm(18));

    // ---- decoder levels 4..1 ----
    float* xbCur = xbB[0];
    for (int lvl = 4; lvl >= 1; --lvl) {
        int nt = cnt[lvl - 1], ns = cnt[lvl];
        int out_p = wid[lvl - 1], in_p = wid[lvl];
        int b0 = 21 + (4 - lvl) * 8;
        float* xbNext = xbB[5 - lvl];

        // y = relu(bn(skip @ l1_W + l1_b))
        gemm(feats[lvl - 1], prm(b0 + 0), prm(b0 + 1), bufPre, nt, out_p, out_p, stream);
        bn_stats_kernel<<<out_p, 256, 0, stream>>>(bufPre, nt, out_p, meanB, rstdB);
        bn_relu_kernel<<<GRD((size_t)nt * out_p, TPB), TPB, 0, stream>>>(
            bufPre, bufY, nt * out_p, out_p, meanB, rstdB, prm(b0 + 2), prm(b0 + 3));

        // z = relu(bn(xb @ l2_W + l2_b))
        gemm(xbCur, prm(b0 + 4), prm(b0 + 5), bufPre, ns, out_p, in_p, stream);
        bn_stats_kernel<<<out_p, 256, 0, stream>>>(bufPre, ns, out_p, meanB, rstdB);
        bn_relu_kernel<<<GRD((size_t)ns * out_p, TPB), TPB, 0, stream>>>(
            bufPre, bufZ, ns * out_p, out_p, meanB, rstdB, prm(b0 + 6), prm(b0 + 7));

        // xb = y + interpolate(z)
        interp_add_kernel<<<GRD((size_t)nt * out_p, TPB), TPB, 0, stream>>>(
            pts[lvl - 1], pts[lvl], itA[lvl - 1], bufZ, bufY, xbNext, nt, out_p);
        xbCur = xbNext;
    }

    // ---- decoder_boundary + boundary head ----
    gemm(xbCur, prm(53), prm(54), bufPre, 16384, 32, 32, stream);
    bn_stats_kernel<<<32, 256, 0, stream>>>(bufPre, 16384, 32, meanB, rstdB);
    bn_relu_kernel<<<GRD((size_t)16384 * 32, TPB), TPB, 0, stream>>>(
        bufPre, h1, 16384 * 32, 32, meanB, rstdB, prm(55), prm(56));
    gemm(h1, prm(57), prm(58), h2, 16384, 32, 32, stream);      // linear only
    gemm(h2, prm(59), prm(60), bufPre, 16384, 32, 32, stream);
    bn_stats_kernel<<<32, 256, 0, stream>>>(bufPre, 16384, 32, meanB, rstdB);
    bn_relu_kernel<<<GRD((size_t)16384 * 32, TPB), TPB, 0, stream>>>(
        bufPre, h1, 16384 * 32, 32, meanB, rstdB, prm(61), prm(62));

    // final logits: pad bd_W2/bd_b2 to N=16 so the GEMM runs the interior path
    pad_head_kernel<<<GRD(528, TPB), TPB, 0, stream>>>(prm(63), prm(64), Wpad, bpad);
    gemm(h1, Wpad, bpad, outPad, 16384, 16, 32, stream);
    extract2_kernel<<<GRD((size_t)16384 * 2, TPB), TPB, 0, stream>>>(
        outPad, (float*)d_out, 16384);   // (16384, 2) logits
}
